// MambaTokenBlock_22282290332046
// MI455X (gfx1250) — matrix operand order repaired and checked by reading, not compiled, not run on hardware
//
#include <hip/hip_runtime.h>
#include <hip/hip_bf16.h>

#define XDIM   1024
#define DI     2048
#define DI2    4096
#define DSTATE 16
#define DCONV  4
#define DTRANK 64
#define NBATCH 2
#define SEQ    2048
#define NTOK   (NBATCH * SEQ)          // 4096 tokens
#define DBCN   (DTRANK + 2 * DSTATE)   // 96

typedef __bf16 bf16_t;
typedef __attribute__((ext_vector_type(16))) __bf16 v16bf;
typedef __attribute__((ext_vector_type(8)))  __bf16 v8bf;
typedef __attribute__((ext_vector_type(8)))  float  v8f;

__device__ __forceinline__ bf16_t f2bf(float f) {
    unsigned u = __float_as_uint(f);
    unsigned r = u + 0x7FFFu + ((u >> 16) & 1u);   // round-to-nearest-even
    unsigned short h = (unsigned short)(r >> 16);
    return __builtin_bit_cast(bf16_t, h);
}

// ---------------------------------------------------------------- converts
__global__ __launch_bounds__(256)
void convert_f32_bf16(const float* __restrict__ s, bf16_t* __restrict__ d, int n) {
    int i = blockIdx.x * 256 + threadIdx.x;
    if (i < n) d[i] = f2bf(s[i]);
}

__global__ __launch_bounds__(256)
void extract_dt_kernel(const float* __restrict__ dbc, bf16_t* __restrict__ dt) {
    int i = blockIdx.x * 256 + threadIdx.x;      // NTOK * DTRANK
    int t = i >> 6;
    int r = i & 63;
    dt[i] = f2bf(dbc[(size_t)t * DBCN + r]);
}

// ---------------------------------------------------------------- layernorm
__global__ __launch_bounds__(256)
void ln_kernel(const float* __restrict__ x, const float* __restrict__ w,
               const float* __restrict__ b, bf16_t* __restrict__ out) {
    int t = blockIdx.x;
    const float4* xr = (const float4*)(x + (size_t)t * XDIM);
    float4 v = xr[threadIdx.x];
    float s  = v.x + v.y + v.z + v.w;
    float ss = v.x * v.x + v.y * v.y + v.z * v.z + v.w * v.w;
    #pragma unroll
    for (int off = 16; off; off >>= 1) {
        s  += __shfl_xor(s,  off, 32);
        ss += __shfl_xor(ss, off, 32);
    }
    __shared__ float sh[16];
    __shared__ float mv[2];
    int wv = threadIdx.x >> 5;
    if ((threadIdx.x & 31) == 0) { sh[wv] = s; sh[8 + wv] = ss; }
    __syncthreads();
    if (threadIdx.x == 0) {
        float a = 0.f, a2 = 0.f;
        #pragma unroll
        for (int i = 0; i < 8; i++) { a += sh[i]; a2 += sh[8 + i]; }
        float mu  = a  * (1.f / XDIM);
        float var = a2 * (1.f / XDIM) - mu * mu;
        mv[0] = mu; mv[1] = rsqrtf(var + 1e-5f);
    }
    __syncthreads();
    float mu = mv[0], rs = mv[1];
    float4 wv4 = ((const float4*)w)[threadIdx.x];
    float4 bv4 = ((const float4*)b)[threadIdx.x];
    bf16_t* o = out + (size_t)t * XDIM + threadIdx.x * 4;
    o[0] = f2bf((v.x - mu) * rs * wv4.x + bv4.x);
    o[1] = f2bf((v.y - mu) * rs * wv4.y + bv4.y);
    o[2] = f2bf((v.z - mu) * rs * wv4.z + bv4.z);
    o[3] = f2bf((v.w - mu) * rs * wv4.w + bv4.w);
}

// ---------------------------------------------------------------- WMMA GEMM
// C[M,N] = A[M,K] @ W[N,K]^T, bf16 inputs, f32 accumulate.
// Block tile 128x128, K-step 32, 8 waves (2x4), each wave 64x32 via 4x2 WMMA.
// Tiles are fetched memory->LDS with GLOBAL_LOAD_ASYNC_TO_LDS_B128 (ASYNCcnt)
// into double-buffered LDS; compute on buffer k overlaps the fill of k+1.
// Out-of-range W rows (N=96 case) are clamped to row N-1: their garbage only
// feeds output columns that the store-side "ncol < N" guard discards.
// EPI: 0 = none, 1 = +bias then softplus, 2 = +resid (residual add)
template<int EPI>
__global__ __launch_bounds__(256)
void gemm_bf16_kernel(const bf16_t* __restrict__ A, const bf16_t* __restrict__ W,
                      float* __restrict__ C, const float* __restrict__ bias,
                      const float* __restrict__ resid, int M, int N, int K) {
    __shared__ __align__(64) bf16_t sA[2][128][32];
    __shared__ __align__(64) bf16_t sB[2][128][32];

    const int tid  = threadIdx.x;
    const int lane = tid & 31;
    const int wave = tid >> 5;
    const int wm   = wave & 1;        // wave row (2)
    const int wn   = wave >> 1;       // wave col (4)
    const int half = lane >> 4;
    const int l16  = lane & 15;
    const int bM   = blockIdx.y * 128;
    const int bN   = blockIdx.x * 128;

    // Issue the async fills for one 128x32 A tile and one 128x32 W tile:
    // 512 16-byte chunks per tile, 2 per thread per tile.
    auto issue_tile = [&](int buf, int k0) {
        #pragma unroll
        for (int i = 0; i < 2; ++i) {
            int c  = tid + i * 256;
            int r  = c >> 2;           // tile row 0..127
            int cc = (c & 3) << 3;     // bf16 col 0,8,16,24
            unsigned long long ga =
                (unsigned long long)(A + (size_t)(bM + r) * K + k0 + cc);
            unsigned la = (unsigned)(uintptr_t)(&sA[buf][r][cc]);
            asm volatile("global_load_async_to_lds_b128 %0, %1, off"
                         :: "v"(la), "v"(ga) : "memory");
            int n = bN + r;
            if (n >= N) n = N - 1;     // clamp; result discarded at store
            unsigned long long gw =
                (unsigned long long)(W + (size_t)n * K + k0 + cc);
            unsigned lb = (unsigned)(uintptr_t)(&sB[buf][r][cc]);
            asm volatile("global_load_async_to_lds_b128 %0, %1, off"
                         :: "v"(lb), "v"(gw) : "memory");
        }
    };

    v8f acc[4][2];
    #pragma unroll
    for (int mi = 0; mi < 4; mi++)
        #pragma unroll
        for (int ni = 0; ni < 2; ni++) acc[mi][ni] = {};

    issue_tile(0, 0);
    asm volatile("s_wait_asynccnt 0x0" ::: "memory");
    __syncthreads();

    int cur = 0;
    for (int k0 = 0; k0 < K; k0 += 32) {
        int k1 = k0 + 32;
        if (k1 < K) issue_tile(cur ^ 1, k1);     // prefetch next slice

        v16bf bfr[2], afr[4];
        #pragma unroll
        for (int ni = 0; ni < 2; ni++)
            bfr[ni] = *(const v16bf*)(&sB[cur][wn * 32 + ni * 16 + l16][half * 16]);
        #pragma unroll
        for (int mi = 0; mi < 4; mi++) {
            const bf16_t* row = &sA[cur][wm * 64 + mi * 16 + l16][0];
            v8bf lo = *(const v8bf*)(row + half * 16);
            v8bf hi = *(const v8bf*)(row + half * 16 + 8);
            afr[mi] = __builtin_shufflevector(lo, hi,
                0, 1, 2, 3, 4, 5, 6, 7, 8, 9, 10, 11, 12, 13, 14, 15);
        }
        #pragma unroll
        for (int mi = 0; mi < 4; mi++)
            #pragma unroll
            for (int ni = 0; ni < 2; ni++)
                acc[mi][ni] = __builtin_amdgcn_wmma_f32_16x16x32_bf16(
                    false, afr[mi], false, bfr[ni], (short)0, acc[mi][ni],
                    false, false);

        asm volatile("s_wait_asynccnt 0x0" ::: "memory");  // next buffer landed
        __syncthreads();                                    // all waves done w/ cur
        cur ^= 1;
    }

    #pragma unroll
    for (int mi = 0; mi < 4; mi++) {
        #pragma unroll
        for (int ni = 0; ni < 2; ni++) {
            int ncol = bN + wn * 32 + ni * 16 + l16;
            if (ncol < N) {
                #pragma unroll
                for (int r = 0; r < 8; r++) {
                    int mrow = bM + wm * 64 + mi * 16 + half * 8 + r;
                    size_t idx = (size_t)mrow * N + ncol;
                    float v = acc[mi][ni][r];
                    if (EPI == 1) {
                        v += bias[ncol];
                        v = (v > 20.f) ? v : log1pf(__expf(v));
                    } else if (EPI == 2) {
                        v += resid[idx];
                    }
                    C[idx] = v;
                }
            }
        }
    }
}

// ---------------------------------------------------------------- conv+SiLU
__global__ __launch_bounds__(256)
void conv_silu_kernel(const float* __restrict__ xz, const float* __restrict__ cw,
                      const float* __restrict__ cb, float* __restrict__ xc,
                      bf16_t* __restrict__ xc_bf) {
    int idx = blockIdx.x * 256 + threadIdx.x;    // NTOK * DI
    int d = idx & (DI - 1);
    int t = idx >> 11;
    int l = t & (SEQ - 1);
    float acc = cb[d];
    #pragma unroll
    for (int j = 0; j < DCONV; j++) {
        int lt = l - (DCONV - 1) + j;
        if (lt >= 0)
            acc += cw[d * DCONV + j] *
                   xz[(size_t)(t - (DCONV - 1) + j) * DI2 + d];
    }
    float s = acc / (1.f + __expf(-acc));        // SiLU
    xc[idx] = s;
    xc_bf[idx] = f2bf(s);
}

// ---------------------------------------------------------------- scan
// One thread per (channel, state); 16-lane shuffle reduction for y = h.C.
// Fuses + u*D and the SiLU(z) gate; writes bf16 for the out_proj GEMM.
__global__ __launch_bounds__(256)
void scan_kernel(const float* __restrict__ delta, const float* __restrict__ xc,
                 const float* __restrict__ dbc, const float* __restrict__ xz,
                 const float* __restrict__ A_log, const float* __restrict__ Dv,
                 bf16_t* __restrict__ ybf) {
    int n = threadIdx.x & 15;                          // state index
    int d = blockIdx.x * 16 + (threadIdx.x >> 4);      // channel
    int b = blockIdx.y;
    float An = -__expf(A_log[d * DSTATE + n]);
    float Dd = Dv[d];
    float h  = 0.f;
    size_t tok0 = (size_t)b * SEQ;
    for (int l = 0; l < SEQ; l++) {
        size_t t  = tok0 + l;
        float dl = delta[t * DI + d];
        float u  = xc[t * DI + d];
        float Bn = dbc[t * DBCN + DTRANK + n];
        float Cn = dbc[t * DBCN + DTRANK + DSTATE + n];
        h = __expf(dl * An) * h + (dl * u) * Bn;
        float y = h * Cn;
        #pragma unroll
        for (int off = 1; off < 16; off <<= 1) y += __shfl_xor(y, off, 32);
        if (n == 0) {
            float z = xz[t * DI2 + DI + d];
            float g = z / (1.f + __expf(-z));
            ybf[t * DI + d] = f2bf((y + u * Dd) * g);
        }
    }
}

// ---------------------------------------------------------------- launch
extern "C" void kernel_launch(void* const* d_in, const int* in_sizes, int n_in,
                              void* d_out, int out_size, void* d_ws, size_t ws_size,
                              hipStream_t stream) {
    (void)in_sizes; (void)n_in; (void)out_size; (void)ws_size;
    const float* x          = (const float*)d_in[0];
    const float* ln_w       = (const float*)d_in[1];
    const float* ln_b       = (const float*)d_in[2];
    const float* in_proj_w  = (const float*)d_in[3];
    const float* conv_w     = (const float*)d_in[4];
    const float* conv_b     = (const float*)d_in[5];
    const float* x_proj_w   = (const float*)d_in[6];
    const float* dt_proj_w  = (const float*)d_in[7];
    const float* dt_proj_b  = (const float*)d_in[8];
    const float* A_log      = (const float*)d_in[9];
    const float* Dvec       = (const float*)d_in[10];
    const float* out_proj_w = (const float*)d_in[11];
    float* out = (float*)d_out;

    char* p = (char*)d_ws;
    auto carve = [&](size_t bytes) -> void* {
        void* r = (void*)p;
        p += (bytes + 255) & ~(size_t)255;
        return r;
    };
    bf16_t* xln_bf   = (bf16_t*)carve((size_t)NTOK * XDIM  * 2);
    bf16_t* w_in_bf  = (bf16_t*)carve((size_t)DI2  * XDIM  * 2);
    bf16_t* w_xp_bf  = (bf16_t*)carve((size_t)DBCN * DI    * 2);
    bf16_t* w_dt_bf  = (bf16_t*)carve((size_t)DI * DTRANK  * 2);
    bf16_t* w_out_bf = (bf16_t*)carve((size_t)XDIM * DI    * 2);
    float*  xz       = (float*) carve((size_t)NTOK * DI2   * 4);
    float*  xc_act   = (float*) carve((size_t)NTOK * DI    * 4);
    bf16_t* xc_bf    = (bf16_t*)carve((size_t)NTOK * DI    * 2);
    float*  dbc      = (float*) carve((size_t)NTOK * DBCN  * 4);
    bf16_t* dt_bf    = (bf16_t*)carve((size_t)NTOK * DTRANK* 2);
    float*  delta    = (float*) carve((size_t)NTOK * DI    * 4);
    bf16_t* y_bf     = (bf16_t*)carve((size_t)NTOK * DI    * 2);

    auto cvt = [&](const float* s, bf16_t* d, int n) {
        convert_f32_bf16<<<(n + 255) / 256, 256, 0, stream>>>(s, d, n);
    };
    cvt(in_proj_w,  w_in_bf,  DI2 * XDIM);
    cvt(x_proj_w,   w_xp_bf,  DBCN * DI);
    cvt(dt_proj_w,  w_dt_bf,  DI * DTRANK);
    cvt(out_proj_w, w_out_bf, XDIM * DI);

    ln_kernel<<<NTOK, 256, 0, stream>>>(x, ln_w, ln_b, xln_bf);

    // xz = x_ln @ in_proj_w^T   (4096 x 4096, K=1024)
    gemm_bf16_kernel<0><<<dim3(DI2 / 128, NTOK / 128), 256, 0, stream>>>(
        xln_bf, w_in_bf, xz, nullptr, nullptr, NTOK, DI2, XDIM);

    conv_silu_kernel<<<(NTOK * DI) / 256, 256, 0, stream>>>(
        xz, conv_w, conv_b, xc_act, xc_bf);

    // dbc = xc @ x_proj_w^T   (4096 x 96, K=2048)
    gemm_bf16_kernel<0><<<dim3(1, NTOK / 128), 256, 0, stream>>>(
        xc_bf, w_xp_bf, dbc, nullptr, nullptr, NTOK, DBCN, DI);

    extract_dt_kernel<<<(NTOK * DTRANK) / 256, 256, 0, stream>>>(dbc, dt_bf);

    // delta = softplus(dt @ dt_proj_w^T + b)   (4096 x 2048, K=64)
    gemm_bf16_kernel<1><<<dim3(DI / 128, NTOK / 128), 256, 0, stream>>>(
        dt_bf, w_dt_bf, delta, dt_proj_b, nullptr, NTOK, DI, DTRANK);

    scan_kernel<<<dim3(DI / 16, NBATCH), 256, 0, stream>>>(
        delta, xc_act, dbc, xz, A_log, Dvec, y_bf);

    // out = x + y @ out_proj_w^T   (4096 x 1024, K=2048)
    gemm_bf16_kernel<2><<<dim3(XDIM / 128, NTOK / 128), 256, 0, stream>>>(
        y_bf, w_out_bf, out, nullptr, x, NTOK, XDIM, DI);
}
